// TransformerBlock_masked_75161927680022
// MI455X (gfx1250) — matrix operand order repaired and checked by reading, not compiled, not run on hardware
//
#include <hip/hip_runtime.h>

// ---------------- problem constants ----------------
constexpr int BB  = 4;
constexpr int NN  = 4096;
constexpr int KNB = 16;
constexpr int DPT = 64;
constexpr int DM  = 128;

// ---------------- vector types ----------------
typedef _Float16 v16h __attribute__((ext_vector_type(16)));
typedef _Float16 h8   __attribute__((ext_vector_type(8)));
typedef float    v8f  __attribute__((ext_vector_type(8)));
typedef unsigned int u32x4 __attribute__((ext_vector_type(4)));
typedef int          i32x4 __attribute__((ext_vector_type(4)));
typedef int          i32x8 __attribute__((ext_vector_type(8)));

union Frag16 { v16h v; h8 h[2]; };

#if defined(__HIP_DEVICE_COMPILE__) && __has_builtin(__builtin_amdgcn_tensor_load_to_lds)
#define USE_TDM 1
#else
#define USE_TDM 0
#endif

// ------------------------------------------------------------------
// TDM: 1-D copy of `elems` f16 elements global->LDS via the Tensor
// Data Mover (D# per CDNA5 ISA 8.3/8.4).  Issue from a single wave.
//   group0: count=1 | lds_addr | global_addr | type=2
//   group1: data_size=2B, tensor_dim0=tile_dim0=elems, dim1=1,
//           tensor_dim0_stride=elems, no pad/iterate/multicast.
// This toolchain exposes the 6-arg builtin:
//   (u32x4 g0, i32x8 g1, i32x4, i32x4, i32x8, i32 cpol)
// ------------------------------------------------------------------
#if USE_TDM
__device__ inline void tdm_load_1d(const _Float16* gsrc, _Float16* ldst,
                                   int elems) {
  const unsigned lds_off = (unsigned)(size_t)(void*)ldst;
  const unsigned long long ga = (unsigned long long)(size_t)gsrc;
  u32x4 g0;
  g0[0] = 1u;                                 // count=1 (valid descriptor)
  g0[1] = lds_off;                            // lds_addr [63:32]
  g0[2] = (unsigned)ga;                       // global_addr [95:64]
  g0[3] = (unsigned)(ga >> 32) | (2u << 30);  // global_addr hi + type=2
  i32x8 g1;
  g1[0] = 0x00010000;                         // data_size = 1 (2 bytes)
  g1[1] = (elems & 0xFFFF) << 16;             // tensor_dim0 [15:0]
  g1[2] = (elems >> 16) | (1 << 16);          // tensor_dim0 hi | tensor_dim1=1
  g1[3] = (elems & 0xFFFF) << 16;             // tile_dim0
  g1[4] = 0;                                  // tile_dim1=0, tile_dim2=0
  g1[5] = elems;                              // tensor_dim0_stride [31:0]
  g1[6] = 0;
  g1[7] = 0;
  const i32x4 z4 = {0, 0, 0, 0};
  const i32x8 z8 = {0, 0, 0, 0, 0, 0, 0, 0};
  __builtin_amdgcn_tensor_load_to_lds(g0, g1, z4, z4, z8, 0);
}
#endif

// Stage a pre-transposed f16 weight into LDS (TDM if available).
__device__ inline void stage_weight(const _Float16* g, _Float16* l, int elems,
                                    int tid, int wave) {
#if USE_TDM
  if (wave == 0) tdm_load_1d(g, l, elems);
#else
  for (int e = tid; e < elems; e += 256) l[e] = g[e];
#endif
}
__device__ inline void stage_wait(int wave) {
#if USE_TDM
  if (wave == 0) __builtin_amdgcn_s_wait_tensorcnt(0);
#endif
}

// ------------------------------------------------------------------
// One 16x16 output tile of A[rows][KD] (row-major f16 LDS) times
// Bt[cols][KD] (weight transposed in LDS), f32 accumulate.
// ------------------------------------------------------------------
template<int KD>
__device__ inline v8f wmma_tile(const _Float16* A, const _Float16* Bt,
                                int rt, int ct, int lane) {
  const int l15 = lane & 15;
  const int hi  = (lane >> 4) & 1;
  const _Float16* arow = A  + (rt * 16 + l15) * KD;
  const _Float16* bcol = Bt + (ct * 16 + l15) * KD;
  v8f acc = {};
#pragma unroll
  for (int k0 = 0; k0 < KD; k0 += 32) {
    Frag16 fa;
    fa.h[0] = *(const h8*)(arow + k0 + hi * 16);
    fa.h[1] = *(const h8*)(arow + k0 + hi * 16 + 8);
    v16h bfr = *(const v16h*)(bcol + k0 + hi * 16);
    acc = __builtin_amdgcn_wmma_f32_16x16x32_f16(
        false, fa.v, false, bfr, (short)0, acc, false, false);
  }
  return acc;
}

__device__ inline float sigmoidf_(float x) { return 1.f / (1.f + __expf(-x)); }

// ==================================================================
// Kernel 0: one-time weight prep: f32 [K][128] -> f16 transposed [128][K].
// Block m handles matrix m.
// ==================================================================
__global__ __launch_bounds__(256) void prep_w_kernel(
    const float* __restrict__ fc1_w, const float* __restrict__ wq,
    const float* __restrict__ wk, const float* __restrict__ wv,
    const float* __restrict__ dw2, const float* __restrict__ gw1,
    const float* __restrict__ gw2, const float* __restrict__ sw1,
    _Float16* __restrict__ wt) {
  const int m = blockIdx.x;
  const float* src; int KD; int off;
  switch (m) {
    case 0: src = fc1_w; KD = 64;  off = 0;      break;
    case 1: src = wq;    KD = 128; off = 8192;   break;
    case 2: src = wk;    KD = 128; off = 24576;  break;
    case 3: src = wv;    KD = 128; off = 40960;  break;
    case 4: src = dw2;   KD = 128; off = 57344;  break;
    case 5: src = gw1;   KD = 128; off = 73728;  break;
    case 6: src = gw2;   KD = 128; off = 90112;  break;
    default: src = sw1;  KD = 128; off = 106496; break;
  }
  const int total = KD * DM;
  for (int e = threadIdx.x; e < total; e += 256) {
    const int k = e >> 7, n = e & 127;
    wt[off + n * KD + k] = (_Float16)src[e];
  }
}

// ==================================================================
// Kernel 1: brute-force KNN (top-16, stable-argsort tie-break).
// ==================================================================
__global__ __launch_bounds__(256) void knn_kernel(const float* __restrict__ xyz,
                                                  int* __restrict__ knn) {
  __shared__ float sx[256 * 3];
  const int tid = threadIdx.x;
  const int b   = blockIdx.y;
  const int n   = blockIdx.x * 256 + tid;
  const float* xb = xyz + (size_t)b * NN * 3;
  const float qx = xb[n * 3 + 0], qy = xb[n * 3 + 1], qz = xb[n * 3 + 2];

  unsigned long long best[KNB];
#pragma unroll
  for (int i = 0; i < KNB; ++i) best[i] = ~0ull;

  for (int t = 0; t < NN; t += 256) {
    __syncthreads();
#pragma unroll
    for (int i = 0; i < 3; ++i) sx[tid + i * 256] = xb[t * 3 + tid + i * 256];
    __syncthreads();
    for (int j = 0; j < 256; ++j) {
      const float dx = qx - sx[j * 3 + 0];
      const float dy = qy - sx[j * 3 + 1];
      const float dz = qz - sx[j * 3 + 2];
      const float d  = dx * dx + dy * dy + dz * dz;
      unsigned long long key =
          ((unsigned long long)__float_as_uint(d) << 32) | (unsigned)(t + j);
      if (key < best[KNB - 1]) {
        unsigned long long cur = key;
#pragma unroll
        for (int p = 0; p < KNB; ++p) {
          const unsigned long long old = best[p];
          const bool lt = cur < old;
          best[p] = lt ? cur : old;
          cur     = lt ? old : cur;
        }
      }
    }
  }
  int* op = knn + ((size_t)b * NN + n) * KNB;
#pragma unroll
  for (int i = 0; i < KNB; ++i) op[i] = (int)(best[i] & 0xffffffffu);
}

// ==================================================================
// Kernel 2: x = features@fc1_w + b ; q/k/v = x @ {wq,wk,wv} (f16 WMMA).
// Weights arrive pre-transposed f16 via TDM.
// ==================================================================
__global__ __launch_bounds__(256) void qkv_kernel(
    const float* __restrict__ features, const _Float16* __restrict__ wt_fc1,
    const float* __restrict__ fc1_b, const _Float16* __restrict__ wt_wq,
    const _Float16* __restrict__ wt_wk, const _Float16* __restrict__ wt_wv,
    _Float16* __restrict__ qf, _Float16* __restrict__ kf,
    _Float16* __restrict__ vf) {
  __shared__ _Float16 sF[256 * DPT];   // 32 KB
  __shared__ _Float16 sW[DM * DM];     // 32 KB (transposed weight [n][k])
  __shared__ _Float16 sX[256 * DM];    // 64 KB
  const int tid  = threadIdx.x;
  const int lane = tid & 31, wave = tid >> 5;
  const int l15 = lane & 15, hi = (lane >> 4) & 1;
  const size_t rowBase = (size_t)blockIdx.x * 256;

  stage_weight(wt_fc1, sW, DPT * DM, tid, wave);     // TDM overlaps with sF fill
  for (int e = tid; e < 256 * DPT; e += 256)
    sF[e] = (_Float16)features[rowBase * DPT + e];
  stage_wait(wave);
  __syncthreads();

  for (int t8 = wave; t8 < 128; t8 += 8) {           // x = F @ W1 + b1
    const int rt = t8 >> 3, ct = t8 & 7;
    v8f acc = wmma_tile<DPT>(sF, sW, rt, ct, lane);
    const int col = ct * 16 + l15;
    const float bias = fc1_b[col];
#pragma unroll
    for (int i = 0; i < 8; ++i)
      sX[(rt * 16 + i + hi * 8) * DM + col] = (_Float16)(acc[i] + bias);
  }
  __syncthreads();

  const _Float16* Ws[3] = {wt_wq, wt_wk, wt_wv};
  _Float16*       Os[3] = {qf, kf, vf};
  for (int m = 0; m < 3; ++m) {
    stage_weight(Ws[m], sW, DM * DM, tid, wave);
    stage_wait(wave);
    __syncthreads();
    _Float16* outp = Os[m];
    for (int t8 = wave; t8 < 128; t8 += 8) {
      const int rt = t8 >> 3, ct = t8 & 7;
      v8f acc = wmma_tile<DM>(sX, sW, rt, ct, lane);
      const int col = ct * 16 + l15;
#pragma unroll
      for (int i = 0; i < 8; ++i) {
        const size_t row = rowBase + rt * 16 + i + hi * 8;
        outp[row * DM + col] = (_Float16)acc[i];
      }
    }
    __syncthreads();
  }
}

// ==================================================================
// Kernel 3: per (b, 8-query tile): pos_enc / attention MLPs with WMMA,
// TDM-staged weights, score via ds_add_f32, softmax over 16 neighbors.
// ==================================================================
__global__ __launch_bounds__(256) void attn_kernel(
    const float* __restrict__ xyz, const float* __restrict__ features,
    const int* __restrict__ gt, const int* __restrict__ knn,
    const _Float16* __restrict__ qf, const _Float16* __restrict__ kf,
    const _Float16* __restrict__ vf,
    const float* __restrict__ delta_w1, const float* __restrict__ delta_b1,
    const _Float16* __restrict__ wt_dw2, const float* __restrict__ delta_b2,
    const _Float16* __restrict__ wt_gw1, const float* __restrict__ gamma_b1,
    const _Float16* __restrict__ wt_gw2, const float* __restrict__ gamma_b2,
    const _Float16* __restrict__ wt_sw1, const float* __restrict__ score_b1,
    const float* __restrict__ score_w2, const float* __restrict__ score_b2,
    const float* __restrict__ fc2_w, const float* __restrict__ fc2_b,
    float* __restrict__ out_res, float* __restrict__ out_gt,
    float* __restrict__ out_cm) {
  __shared__ _Float16 sX[128 * DM];   // h1 -> diff -> attn
  __shared__ _Float16 sY[128 * DM];   // pos_enc -> attn_h
  __shared__ _Float16 sPV[128 * DM];  // v + pos_enc
  __shared__ _Float16 sW[DM * DM];    // current weight (transposed)
  __shared__ float sRes[8 * DM];
  __shared__ float sScore[128];
  __shared__ float sCM[128];
  __shared__ int   sIdx[128];

  const int tid  = threadIdx.x;
  const int lane = tid & 31, wave = tid >> 5;
  const int l15 = lane & 15, hi = (lane >> 4) & 1;
  const int b     = blockIdx.x >> 9;
  const int ntile = blockIdx.x & 511;
  const int nbase = ntile * 8;
  const size_t pbase = (size_t)b * NN;

  // ---- step 0: TDM(delta_w2), neighbor idx, prefetch gathered rows ----
  stage_weight(wt_dw2, sW, DM * DM, tid, wave);
  if (tid < 128) {
    const int n_l = tid >> 4, kk2 = tid & 15;
    const int gi = knn[(pbase + nbase + n_l) * KNB + kk2];
    sIdx[tid]   = gi;
    sScore[tid] = 0.f;
    __builtin_prefetch((const void*)(kf + (pbase + gi) * DM), 0, 1);
    __builtin_prefetch((const void*)(vf + (pbase + gi) * DM), 0, 1);
  }
  __syncthreads();

  // ---- step 1: h1 = relu(rel_xyz @ delta_w1 + b1) -> sX (K=3, VALU) ----
  {
    const int r = tid >> 1, ch = tid & 1;
    const int n = nbase + (r >> 4);
    const int gi = sIdx[r];
    const float rx = xyz[(pbase + n) * 3 + 0] - xyz[(pbase + gi) * 3 + 0];
    const float ry = xyz[(pbase + n) * 3 + 1] - xyz[(pbase + gi) * 3 + 1];
    const float rz = xyz[(pbase + n) * 3 + 2] - xyz[(pbase + gi) * 3 + 2];
    const int c0 = ch * 64;
    for (int c = c0; c < c0 + 64; ++c) {
      const float h = rx * delta_w1[c] + ry * delta_w1[DM + c] +
                      rz * delta_w1[2 * DM + c] + delta_b1[c];
      sX[r * DM + c] = (_Float16)fmaxf(h, 0.f);
    }
  }
  stage_wait(wave);
  __syncthreads();

  // ---- step 2: pos_enc = h1 @ delta_w2 + b2 -> sY (WMMA) ----
  for (int t8 = wave; t8 < 64; t8 += 8) {
    const int rt = t8 >> 3, ct = t8 & 7;
    v8f acc = wmma_tile<DM>(sX, sW, rt, ct, lane);
    const int col = ct * 16 + l15;
    const float bias = delta_b2[col];
#pragma unroll
    for (int i = 0; i < 8; ++i)
      sY[(rt * 16 + i + hi * 8) * DM + col] = (_Float16)(acc[i] + bias);
  }
  __syncthreads();

  // ---- step 3: TDM(gamma_w1); diff -> sX, posv -> sPV (packed f16) ----
  stage_weight(wt_gw1, sW, DM * DM, tid, wave);
  {
    const int r = tid >> 1, ch = tid & 1;
    const int n = nbase + (r >> 4);
    const int gi = sIdx[r];
    const _Float16* qrow = qf + (pbase + n) * DM;
    const _Float16* krow = kf + (pbase + gi) * DM;
    const _Float16* vrow = vf + (pbase + gi) * DM;
    const int c0 = ch * 64;
#pragma unroll
    for (int cc = 0; cc < 8; ++cc) {
      const int c = c0 + cc * 8;
      const h8 qv = *(const h8*)(qrow + c);
      const h8 kv = *(const h8*)(krow + c);
      const h8 vv = *(const h8*)(vrow + c);
      const h8 pe = *(const h8*)(&sY[r * DM + c]);
      *(h8*)(&sPV[r * DM + c]) = vv + pe;
      *(h8*)(&sX[r * DM + c])  = qv - kv + pe;
    }
  }
  stage_wait(wave);
  __syncthreads();

  // ---- step 4: attn_h = relu(diff @ gamma_w1 + b1) -> sY (WMMA) ----
  for (int t8 = wave; t8 < 64; t8 += 8) {
    const int rt = t8 >> 3, ct = t8 & 7;
    v8f acc = wmma_tile<DM>(sX, sW, rt, ct, lane);
    const int col = ct * 16 + l15;
    const float bias = gamma_b1[col];
#pragma unroll
    for (int i = 0; i < 8; ++i)
      sY[(rt * 16 + i + hi * 8) * DM + col] =
          (_Float16)fmaxf(acc[i] + bias, 0.f);
  }
  __syncthreads();

  // ---- step 5: TDM(score_w1) ----
  stage_weight(wt_sw1, sW, DM * DM, tid, wave);
  stage_wait(wave);
  __syncthreads();

  // ---- step 6: score[row] += relu(diff@score_w1+b1)[col]*score_w2[col] ----
  for (int t8 = wave; t8 < 64; t8 += 8) {
    const int rt = t8 >> 3, ct = t8 & 7;
    v8f acc = wmma_tile<DM>(sX, sW, rt, ct, lane);
    const int col = ct * 16 + l15;
    const float b1v = score_b1[col];
    const float w2v = score_w2[col];
#pragma unroll
    for (int i = 0; i < 8; ++i) {
      const int row = rt * 16 + i + hi * 8;
      atomicAdd(&sScore[row], fmaxf(acc[i] + b1v, 0.f) * w2v);
    }
  }
  __syncthreads();

  // ---- step 7: TDM(gamma_w2); class_map = sigmoid(score+b2); gt_mask ----
  stage_weight(wt_gw2, sW, DM * DM, tid, wave);
  if (tid < 128) {
    const int n_l = tid >> 4, kk2 = tid & 15;
    const int n = nbase + n_l;
    const int gi = sIdx[tid];
    const float cmv = sigmoidf_(sScore[tid] + score_b2[0]);
    sCM[tid] = cmv;
    out_cm[(pbase + n) * KNB + kk2] = cmv;
    out_gt[(pbase + n) * KNB + kk2] =
        (gt[pbase + gi] == gt[pbase + n]) ? 1.f : 0.f;
  }
  stage_wait(wave);
  __syncthreads();

  // ---- step 8: attn = (attn_h @ gamma_w2 + b2) * class_map -> sX ----
  for (int t8 = wave; t8 < 64; t8 += 8) {
    const int rt = t8 >> 3, ct = t8 & 7;
    v8f acc = wmma_tile<DM>(sY, sW, rt, ct, lane);
    const int col = ct * 16 + l15;
    const float b2v = gamma_b2[col];
#pragma unroll
    for (int i = 0; i < 8; ++i) {
      const int row = rt * 16 + i + hi * 8;
      sX[row * DM + col] = (_Float16)((acc[i] + b2v) * sCM[row]);
    }
  }
  __syncthreads();

  // ---- step 9: softmax over 16 neighbors (scale 1/sqrt(128)) + weighted sum
  {
    const int n_l = tid >> 5;
    const int f0  = (tid & 31) * 4;
    const float inv = 0.08838834764831845f;
    for (int fo = 0; fo < 4; ++fo) {
      const int f = f0 + fo;
      float m = -1e30f;
#pragma unroll
      for (int kk2 = 0; kk2 < KNB; ++kk2)
        m = fmaxf(m, (float)sX[(n_l * 16 + kk2) * DM + f]);
      float s = 0.f, a = 0.f;
#pragma unroll
      for (int kk2 = 0; kk2 < KNB; ++kk2) {
        const float z = (float)sX[(n_l * 16 + kk2) * DM + f];
        const float e2 = __expf((z - m) * inv);
        s += e2;
        a += e2 * (float)sPV[(n_l * 16 + kk2) * DM + f];
      }
      sRes[n_l * DM + f] = a / s;
    }
  }
  __syncthreads();

  // ---- step 10: out = res @ fc2_w + fc2_b + features (residual) ----
  for (int o = tid; o < 8 * DPT; o += 256) {
    const int n_l = o >> 6, j = o & 63;
    float sum = fc2_b[j];
    for (int c = 0; c < DM; ++c) sum += sRes[n_l * DM + c] * fc2_w[c * DPT + j];
    const int n = nbase + n_l;
    sum += features[(pbase + n) * DPT + j];
    out_res[(pbase + n) * DPT + j] = sum;
  }
}

// ==================================================================
extern "C" void kernel_launch(void* const* d_in, const int* in_sizes, int n_in,
                              void* d_out, int out_size, void* d_ws,
                              size_t ws_size, hipStream_t stream) {
  const float* xyz       = (const float*)d_in[0];
  const float* features  = (const float*)d_in[1];
  const int*   gt        = (const int*)d_in[2];
  const float* fc1_w     = (const float*)d_in[3];
  const float* fc1_b     = (const float*)d_in[4];
  const float* fc2_w     = (const float*)d_in[5];
  const float* fc2_b     = (const float*)d_in[6];
  const float* delta_w1  = (const float*)d_in[7];
  const float* delta_b1  = (const float*)d_in[8];
  const float* delta_w2  = (const float*)d_in[9];
  const float* delta_b2  = (const float*)d_in[10];
  const float* gamma_w1  = (const float*)d_in[11];
  const float* gamma_b1  = (const float*)d_in[12];
  const float* gamma_w2  = (const float*)d_in[13];
  const float* gamma_b2  = (const float*)d_in[14];
  const float* score_w1  = (const float*)d_in[15];
  const float* score_b1  = (const float*)d_in[16];
  const float* score_w2  = (const float*)d_in[17];
  const float* score_b2  = (const float*)d_in[18];
  const float* wq        = (const float*)d_in[19];
  const float* wk        = (const float*)d_in[20];
  const float* wv        = (const float*)d_in[21];

  char* ws = (char*)d_ws;
  int* knn = (int*)ws;                                        // 1 MB
  _Float16* qfp = (_Float16*)(ws + (size_t)BB * NN * KNB * sizeof(int));
  _Float16* kfp = qfp + (size_t)BB * NN * DM;                 // 4 MB each
  _Float16* vfp = kfp + (size_t)BB * NN * DM;
  _Float16* wt  = vfp + (size_t)BB * NN * DM;                 // 240 KB weights

  const _Float16* wt_fc1 = wt + 0;
  const _Float16* wt_wq  = wt + 8192;
  const _Float16* wt_wk  = wt + 24576;
  const _Float16* wt_wv  = wt + 40960;
  const _Float16* wt_dw2 = wt + 57344;
  const _Float16* wt_gw1 = wt + 73728;
  const _Float16* wt_gw2 = wt + 90112;
  const _Float16* wt_sw1 = wt + 106496;

  float* out_res = (float*)d_out;                             // [B,N,64]
  float* out_gt  = out_res + (size_t)BB * NN * DPT;           // [B,N,K]
  float* out_cm  = out_gt + (size_t)BB * NN * KNB;            // [B,N,K]

  prep_w_kernel<<<8, 256, 0, stream>>>(fc1_w, wq, wk, wv, delta_w2, gamma_w1,
                                       gamma_w2, score_w1, wt);
  knn_kernel<<<dim3(NN / 256, BB), 256, 0, stream>>>(xyz, knn);
  qkv_kernel<<<(BB * NN) / 256, 256, 0, stream>>>(features, wt_fc1, fc1_b,
                                                  wt_wq, wt_wk, wt_wv, qfp,
                                                  kfp, vfp);
  attn_kernel<<<BB * (NN / 8), 256, 0, stream>>>(
      xyz, features, gt, knn, qfp, kfp, vfp, delta_w1, delta_b1, wt_dw2,
      delta_b2, wt_gw1, gamma_b1, wt_gw2, gamma_b2, wt_sw1, score_b1, score_w2,
      score_b2, fc2_w, fc2_b, out_res, out_gt, out_cm);
}
